// LSSViewTransformer_81372450390079
// MI455X (gfx1250) — matrix operand order repaired and checked by reading, not compiled, not run on hardware
//
#include <hip/hip_runtime.h>
#include <hip/hip_bf16.h>
#include <cstddef>
#include <cstdint>

// ---------------- problem constants (from reference) ----------------
#define NB      4
#define NCAM    6
#define NIMG    (NB * NCAM)      // 24
#define CAMC    64
#define DD      59               // depth bins
#define OUTC    64
#define OCPAD   128              // 123 padded to 128 for WMMA
#define FH      16
#define FW      44
#define PIX     (FH * FW)        // 704
#define NX_     128
#define NY_     128
#define NVOX    (NX_ * NY_)      // 16384

typedef __attribute__((ext_vector_type(16))) _Float16 v16h;
typedef __attribute__((ext_vector_type(8)))  float    v8f;
typedef unsigned int u32x4 __attribute__((ext_vector_type(4)));
typedef int          i32x4 __attribute__((ext_vector_type(4)));
typedef int          i32x8 __attribute__((ext_vector_type(8)));

// ---------------- kernel 1: W f32 -> f16 (padded to 128 rows) + padded bias ----
__global__ void lss_prep_wb(const float* __restrict__ w, const float* __restrict__ b,
                            _Float16* __restrict__ wp, float* __restrict__ bpad) {
    int i = blockIdx.x * blockDim.x + threadIdx.x;          // 0 .. 128*64-1
    if (i >= OCPAD * CAMC) return;
    int o = i >> 6;
    int c = i & 63;
    wp[i] = (o < DD + OUTC) ? (_Float16)w[o * CAMC + c] : (_Float16)0.0f;
    if (i < OCPAD) bpad[i] = (i < DD + OUTC) ? b[i] : 0.0f;
}

// ---------------- kernel 2: per-image geometry transforms ----------------
__device__ __forceinline__ void inv3(const float* m, float* inv) {
    float a = m[0], b = m[1], c = m[2];
    float d = m[3], e = m[4], f = m[5];
    float g = m[6], h = m[7], i = m[8];
    float A =  (e * i - f * h);
    float B = -(d * i - f * g);
    float C =  (d * h - e * g);
    float r = 1.0f / (a * A + b * B + c * C);
    inv[0] = A * r;               inv[1] = -(b * i - c * h) * r; inv[2] =  (b * f - c * e) * r;
    inv[3] = B * r;               inv[4] =  (a * i - c * g) * r; inv[5] = -(a * f - c * d) * r;
    inv[6] = C * r;               inv[7] = -(a * h - b * g) * r; inv[8] =  (a * e - b * d) * r;
}

__global__ void lss_prep_geom(const float* __restrict__ rots,
                              const float* __restrict__ trans,
                              const float* __restrict__ intrins,
                              const float* __restrict__ post_rots,
                              const float* __restrict__ post_trans,
                              float* __restrict__ gtx) {
    int img = blockIdx.x * blockDim.x + threadIdx.x;
    if (img >= NIMG) return;
    float M1[9], Kinv[9], comb[9];
    inv3(post_rots + img * 9, M1);
    inv3(intrins   + img * 9, Kinv);
    const float* R = rots + img * 9;
    for (int r = 0; r < 3; ++r)
        for (int c = 0; c < 3; ++c) {
            float s = 0.0f;
            for (int k = 0; k < 3; ++k) s += R[r * 3 + k] * Kinv[k * 3 + c];
            comb[r * 3 + c] = s;
        }
    float* o = gtx + img * 24;
    for (int i = 0; i < 9; ++i) o[i]     = M1[i];
    for (int i = 0; i < 9; ++i) o[9 + i] = comb[i];
    o[18] = post_trans[img * 3 + 0];
    o[19] = post_trans[img * 3 + 1];
    o[20] = post_trans[img * 3 + 2];
    o[21] = trans[img * 3 + 0];
    o[22] = trans[img * 3 + 1];
    o[23] = trans[img * 3 + 2];
}

// ---------------- kernel 3: zero the interleaved BEV accumulator ----------------
__global__ void lss_zero(float* __restrict__ p, int n) {
    int i = blockIdx.x * blockDim.x + threadIdx.x;
    if (i < n) p[i] = 0.0f;
}

// ---------------- kernel 4: voxel rank per (img, d, pixel) ----------------
__global__ void lss_rank(const float* __restrict__ gtx, int* __restrict__ ranks) {
    int idx = blockIdx.x * blockDim.x + threadIdx.x;       // img*59*704 + d*704 + pix
    if (idx >= NIMG * DD * PIX) return;
    int pix = idx % PIX;
    int t   = idx / PIX;
    int d   = t % DD;
    int img = t / DD;
    int h = pix / FW, w = pix % FW;

    const float* g = gtx + img * 24;
    float xf = (float)w * (703.0f / 43.0f);                // linspace(0, IMG_W-1, FW)
    float yf = (float)h * 17.0f;                           // linspace(0, IMG_H-1, FH)
    float df = 1.0f + (float)d;                            // arange(1, 60)

    float p0 = xf - g[18], p1 = yf - g[19], p2 = df - g[20];
    float q0 = g[0] * p0 + g[1] * p1 + g[2] * p2;
    float q1 = g[3] * p0 + g[4] * p1 + g[5] * p2;
    float q2 = g[6] * p0 + g[7] * p1 + g[8] * p2;
    q0 *= q2; q1 *= q2;                                    // un-normalize by depth
    float r0 = g[9]  * q0 + g[10] * q1 + g[11] * q2 + g[21];
    float r1 = g[12] * q0 + g[13] * q1 + g[14] * q2 + g[22];
    float r2 = g[15] * q0 + g[16] * q1 + g[17] * q2 + g[23];

    // trunc semantics to match jnp.trunc (toward zero), then bounds filter
    int gx = (int)truncf((r0 + 51.2f) / 0.8f);
    int gy = (int)truncf((r1 + 51.2f) / 0.8f);
    int gz = (int)truncf((r2 + 10.0f) / 20.0f);
    bool kept = (gx >= 0) && (gx < NX_) && (gy >= 0) && (gy < NY_) && (gz == 0);
    ranks[idx] = kept ? (gy * NX_ + gx) : -1;
}

// ---------------- TDM descriptor helper (2D tile: 16 px wide x 64 ch tall) -----
__device__ __forceinline__ void tdm_load_tile(const float* gsrc, unsigned lds_off) {
    uint64_t ga = (uint64_t)(uintptr_t)gsrc;               // byte address of tile start
    u32x4 g0;
    g0[0] = 1u;                                            // count=1, user mode, no gather
    g0[1] = lds_off;                                       // LDS dest byte address
    g0[2] = (unsigned)(ga & 0xffffffffu);                  // global_addr[31:0]
    g0[3] = (unsigned)((ga >> 32) & 0x01ffffffu)           // global_addr[56:32]
          | (2u << 30);                                    // type = 2 ("image")
    i32x8 g1;
    g1[0] = (int)(2u << 16);                               // data_size = 2 -> 4 bytes
    g1[1] = (int)(((unsigned)PIX & 0xffffu) << 16);        // tensor_dim0[15:0] @ [63:48]
    g1[2] = (int)(((unsigned)PIX >> 16) | (64u << 16));    // dim0[31:16] | tensor_dim1[15:0]
    g1[3] = (int)(16u << 16);                              // tensor_dim1[31:16]=0 | tile_dim0=16
    g1[4] = 64;                                            // tile_dim1=64, tile_dim2=0
    g1[5] = PIX;                                           // tensor_dim0_stride[31:0] = 704
    g1[6] = 0;                                             // stride hi + dim1_stride lo
    g1[7] = 0;
    i32x4 gz4 = {0, 0, 0, 0};
#if __has_include(<hip/amd_detail/amd_gfx1250_TDM.h>)
    i32x8 gz8 = {0, 0, 0, 0, 0, 0, 0, 0};
    __builtin_amdgcn_tensor_load_to_lds(g0, g1, gz4, gz4, gz8, 0);   // 6-arg toolchain
#else
    __builtin_amdgcn_tensor_load_to_lds(g0, g1, gz4, gz4, 0);        // 5-arg toolchain
#endif
}

// ---------------- kernel 5: 1x1 conv via TDM staging + WMMA f16->f32 -----------
// Y[128 x 704] = Wp[128 x 64] * X[64 x 704] per image, + bias on store.
// Block = 256 threads = 8 waves, one (img, 16-pixel tile) per block.
// Wave 0 issues one TENSOR_LOAD_TO_LDS for the shared X tile (64 rows x 16 px,
// f32, row-major in LDS); all 8 waves consume it for their B fragments.
__global__ __launch_bounds__(256) void lss_conv(const float* __restrict__ x,
                                                const _Float16* __restrict__ wp,
                                                const float* __restrict__ bpad,
                                                float* __restrict__ y_ws) {
    int blk  = blockIdx.x;               // 0 .. NIMG*44-1
    int img  = blk / FW;                 // 44 pixel tiles of 16 per image
    int pix0 = (blk % FW) * 16;
    const float* X = x + (size_t)img * CAMC * PIX;   // X[c][pix]

    __shared__ float xs32[CAMC * 16];    // tile, row-major: xs32[c*16 + n]

    if (threadIdx.x < 32) {              // one TDM issue per workgroup (wave 0)
        tdm_load_tile(X + pix0, (unsigned)(uintptr_t)&xs32[0]);
        __builtin_amdgcn_s_wait_tensorcnt(0);
    }
    __syncthreads();

    int t     = threadIdx.x;
    int wave  = t >> 5;
    int lane  = t & 31;
    int m0    = wave * 16;
    int mrow  = lane & 15;               // M (A), N (B/C/D) index for this lane
    int khalf = lane >> 4;               // which K-half this lane holds

    const _Float16* wrow = wp + (size_t)(m0 + mrow) * CAMC;
    __builtin_prefetch(wrow, 0, 0);      // global_prefetch_b8

    v8f acc = {};
    #pragma unroll
    for (int ks = 0; ks < 2; ++ks) {     // K = 64 -> two 16x16x32 steps
        v16h a, b;
        #pragma unroll
        for (int j = 0; j < 8; ++j) {
            // A (16x32 f16): VGPR j holds K = (j&3)*2 + (j>=4?16:0) + khalf*8 (+1)
            int ak = ((j & 3) * 2) + ((j >= 4) ? 16 : 0) + khalf * 8 + ks * 32;
            a[2 * j]     = wrow[ak];
            a[2 * j + 1] = wrow[ak + 1];
            // B (32x16 f16): VGPR j holds K = (j&3)*2 + (j>=4?16:0) + khalf*8 (+1), N = lane&15
            int bk = ((j & 3) * 2) + ((j >= 4) ? 16 : 0) + khalf * 8 + ks * 32;
            b[2 * j]     = (_Float16)xs32[bk * 16 + mrow];
            b[2 * j + 1] = (_Float16)xs32[(bk + 1) * 16 + mrow];
        }
        acc = __builtin_amdgcn_wmma_f32_16x16x32_f16(
            /*neg_a=*/false, a, /*neg_b=*/false, b,
            /*c_mod=*/(short)0, acc, /*reuse_a=*/false, /*reuse_b=*/false);
    }

    // C/D layout: VGPR r -> row m0 + r + khalf*8, col pix0 + (lane&15)
    float* Y = y_ws + (size_t)img * OCPAD * PIX;
    #pragma unroll
    for (int r = 0; r < 8; ++r) {
        int o = m0 + r + khalf * 8;
        Y[o * PIX + pix0 + mrow] = acc[r] + bpad[o];   // bpad zero-padded to 128
    }
}

// ---------------- kernel 6: softmax + splat (atomic scatter-add) ----------------
// One 64-thread block per (img, pixel). Thread c owns context channel c.
// BEV accumulator layout: bev[b][vox][c] so the 64 channel-atomics of one
// point are one contiguous 256B burst.
__global__ __launch_bounds__(64) void lss_scatter(const float* __restrict__ y_ws,
                                                  const int* __restrict__ ranks,
                                                  float* __restrict__ bev) {
    int blk = blockIdx.x;                // img*704 + pix
    int img = blk / PIX;
    int pix = blk % PIX;
    int b   = img / NCAM;

    const float* Y = y_ws + (size_t)img * OCPAD * PIX + pix;
    __shared__ float dprob[DD];
    __shared__ float sinv;

    int t = threadIdx.x;
    if (t < DD) dprob[t] = Y[t * PIX];
    __syncthreads();
    if (t == 0) {
        float mx = dprob[0];
        for (int i = 1; i < DD; ++i) mx = fmaxf(mx, dprob[i]);
        float s = 0.0f;
        for (int i = 0; i < DD; ++i) {
            float e = expf(dprob[i] - mx);
            dprob[i] = e;
            s += e;
        }
        sinv = 1.0f / s;
    }
    __syncthreads();

    float inv_s = sinv;
    float ctx   = Y[(DD + t) * PIX];     // context channel t (o = 59..122)
    const int* rk = ranks + (size_t)img * DD * PIX + pix;
    float* bevb   = bev + (size_t)b * NVOX * OUTC;

    for (int d = 0; d < DD; ++d) {
        int vox = rk[d * PIX];
        if (vox >= 0) {
            atomicAdd(&bevb[(size_t)vox * OUTC + t], dprob[d] * inv_s * ctx);
        }
    }
}

// ---------------- kernel 7: transpose to reference output layout ----------------
// out[b][c][y][x] = bev[b][y*128+x][c]
__global__ void lss_finalize(const float* __restrict__ bev, float* __restrict__ out) {
    int i = blockIdx.x * blockDim.x + threadIdx.x;
    if (i >= NB * OUTC * NVOX) return;
    int yx = i % NVOX;
    int t  = i / NVOX;
    int c  = t % OUTC;
    int b  = t / OUTC;
    out[i] = bev[((size_t)b * NVOX + yx) * OUTC + c];
}

// ---------------- launcher ----------------
extern "C" void kernel_launch(void* const* d_in, const int* in_sizes, int n_in,
                              void* d_out, int out_size, void* d_ws, size_t ws_size,
                              hipStream_t stream) {
    const float* x          = (const float*)d_in[0];
    const float* rots       = (const float*)d_in[1];
    const float* trans      = (const float*)d_in[2];
    const float* intrins    = (const float*)d_in[3];
    const float* post_rots  = (const float*)d_in[4];
    const float* post_trans = (const float*)d_in[5];
    const float* w_depth    = (const float*)d_in[6];
    const float* b_depth    = (const float*)d_in[7];
    float* out = (float*)d_out;

    // workspace carve-up (all 4B aligned)
    char* ws = (char*)d_ws;
    size_t off = 0;
    _Float16* wp   = (_Float16*)(ws + off); off += (size_t)OCPAD * CAMC * 2;       // 16 KB
    float*    bpad = (float*)   (ws + off); off += (size_t)OCPAD * 4;              // 512 B
    float*    y_w  = (float*)   (ws + off); off += (size_t)NIMG * OCPAD * PIX * 4; // 8.65 MB
    int*      rks  = (int*)     (ws + off); off += (size_t)NIMG * DD * PIX * 4;    // 3.99 MB
    float*    gtx  = (float*)   (ws + off); off += (size_t)NIMG * 24 * 4;          // 2.3 KB
    float*    bev  = (float*)   (ws + off); off += (size_t)NB * NVOX * OUTC * 4;   // 16.8 MB

    lss_prep_wb  <<<(OCPAD * CAMC + 255) / 256, 256, 0, stream>>>(w_depth, b_depth, wp, bpad);
    lss_prep_geom<<<1, 32, 0, stream>>>(rots, trans, intrins, post_rots, post_trans, gtx);
    lss_zero     <<<(NB * NVOX * OUTC + 255) / 256, 256, 0, stream>>>(bev, NB * NVOX * OUTC);
    lss_rank     <<<(NIMG * DD * PIX + 255) / 256, 256, 0, stream>>>(gtx, rks);
    lss_conv     <<<NIMG * FW, 256, 0, stream>>>(x, wp, bpad, y_w);
    lss_scatter  <<<NIMG * PIX, 64, 0, stream>>>(y_w, rks, bev);
    lss_finalize <<<(NB * OUTC * NVOX + 255) / 256, 256, 0, stream>>>(bev, out);
}
